// ClusteredAttention_3040836845885
// MI455X (gfx1250) — matrix operand order repaired and checked by reading, not compiled, not run on hardware
//
#include <hip/hip_runtime.h>
#include <hip/hip_bf16.h>
#include <math.h>

#define BB   2
#define LL   4096
#define HID  1024
#define HH   16
#define DD   64
#define CC   256
#define BHD  (BB*HH)
#define NEGV (-1.0e9f)

typedef __attribute__((ext_vector_type(8)))  _Float16 v8h;
typedef __attribute__((ext_vector_type(16))) _Float16 v16h;
typedef __attribute__((ext_vector_type(8)))  float    v8f;
typedef int v4i_vs __attribute__((vector_size(4 * sizeof(int))));

// ---------------------------------------------------------------------------
// CDNA5 async global->LDS copy (ASYNCcnt path), with safe fallback
// Builtin signature (from hipcc diagnostic): (int4 __device__*, int4 __shared__*, imm, imm)
// ---------------------------------------------------------------------------
#if defined(__AMDGCN__) && __has_builtin(__builtin_amdgcn_global_load_async_to_lds_b128)
#define HAVE_ASYNC_LDS 1
#else
#define HAVE_ASYNC_LDS 0
#endif

__device__ __forceinline__ void async_copy16(const _Float16* g, _Float16* l) {
#if HAVE_ASYNC_LDS
  __builtin_amdgcn_global_load_async_to_lds_b128(
      (__attribute__((address_space(1))) v4i_vs*)g,
      (__attribute__((address_space(3))) v4i_vs*)l, 0, 0);
#else
  *(v8h*)l = *(const v8h*)g;
#endif
}
__device__ __forceinline__ void async_fence() {
#if HAVE_ASYNC_LDS
#if __has_builtin(__builtin_amdgcn_s_wait_asynccnt)
  __builtin_amdgcn_s_wait_asynccnt(0);
#else
  asm volatile("s_wait_asynccnt 0x0" ::: "memory");
#endif
#endif
}

__device__ __forceinline__ v16h cat8(v8h lo, v8h hi) {
  v16h r;
#pragma unroll
  for (int i = 0; i < 8; ++i) { r[i] = lo[i]; r[i + 8] = hi[i]; }
  return r;
}
// A fragment (16x32 f16): lane half 0 holds K 0-7 & 16-23, half 1 holds 8-15 & 24-31
__device__ __forceinline__ v16h lda_frag(const _Float16* p, int hf) {
  return cat8(*(const v8h*)(p + hf * 8), *(const v8h*)(p + hf * 8 + 16));
}
// B fragment from Bt stored [N][K]: lane half 0 holds K 0-7 & 16-23, half 1 holds 8-15 & 24-31
__device__ __forceinline__ v16h ldb_frag(const _Float16* p, int hf) {
  return cat8(*(const v8h*)(p + hf * 8), *(const v8h*)(p + hf * 8 + 16));
}
__device__ __forceinline__ v8f wmma16(v16h a, v16h b, v8f c) {
  return __builtin_amdgcn_wmma_f32_16x16x32_f16(false, a, false, b, (short)0, c, false, false);
}

// ---------------------------------------------------------------------------
// Generic 128x128 WMMA GEMM, double-buffered async K slabs:
//   C = A[M][K] * Bt[N][K]^T + bias
// MODE 0: write f16 permuted [B,H,L,D] (QKV projections)
// MODE 1: write f32 [M][N] (final output projection)
// ---------------------------------------------------------------------------
template <int MODE>
__global__ __launch_bounds__(256) void gemm_kernel(
    const _Float16* __restrict__ A, const _Float16* __restrict__ Bt,
    const float* __restrict__ bias, void* __restrict__ outp,
    int M, int N, int K) {
  __shared__ _Float16 As[2][128][40];   // pitch 40 halfs = 80B (16B aligned)
  __shared__ _Float16 Bs[2][128][40];
  const int t = threadIdx.x;
  const int warp = t >> 5, lane = t & 31, hf = lane >> 4, ln = lane & 15;
  const int m0 = (warp & 3) * 32;    // 2 M tiles per wave
  const int n0 = (warp >> 2) * 64;   // 4 N tiles per wave
  const int mBase = blockIdx.y * 128, nBase = blockIdx.x * 128;

  v8f acc[2][4];
#pragma unroll
  for (int i = 0; i < 2; ++i)
#pragma unroll
    for (int j = 0; j < 4; ++j)
#pragma unroll
      for (int r = 0; r < 8; ++r) acc[i][j][r] = 0.f;

  const int arow = t >> 1, aseg = (t & 1) << 4;
  auto issue_tile = [&](int buf, int kb) {
    const _Float16* gA = A  + (size_t)(mBase + arow) * K + kb + aseg;
    const _Float16* gB = Bt + (size_t)(nBase + arow) * K + kb + aseg;
    async_copy16(gA,     &As[buf][arow][aseg]);
    async_copy16(gA + 8, &As[buf][arow][aseg + 8]);
    async_copy16(gB,     &Bs[buf][arow][aseg]);
    async_copy16(gB + 8, &Bs[buf][arow][aseg + 8]);
  };

  issue_tile(0, 0);
  async_fence();
  __syncthreads();

  int cur = 0;
  for (int kb = 0; kb < K; kb += 32) {
    if (kb + 32 < K) issue_tile(cur ^ 1, kb + 32);  // overlap DMA with WMMA
    v16h af[2], bf[4];
#pragma unroll
    for (int i = 0; i < 2; ++i) af[i] = lda_frag(&As[cur][m0 + i * 16 + ln][0], hf);
#pragma unroll
    for (int j = 0; j < 4; ++j) bf[j] = ldb_frag(&Bs[cur][n0 + j * 16 + ln][0], hf);
#pragma unroll
    for (int i = 0; i < 2; ++i)
#pragma unroll
      for (int j = 0; j < 4; ++j) acc[i][j] = wmma16(af[i], bf[j], acc[i][j]);
    async_fence();
    __syncthreads();
    cur ^= 1;
  }
#pragma unroll
  for (int i = 0; i < 2; ++i) {
#pragma unroll
    for (int j = 0; j < 4; ++j) {
      const int gn  = nBase + n0 + j * 16 + ln;
      const int gmb = mBase + m0 + i * 16 + hf * 8;
      const float bv = bias[gn];
#pragma unroll
      for (int r = 0; r < 8; ++r) {
        const int gm = gmb + r;
        const float v = acc[i][j][r] + bv;
        if (MODE == 0) {
          _Float16* O = (_Float16*)outp;
          const int b = gm >> 12, l = gm & (LL - 1);
          const int h = gn >> 6,  d = gn & (DD - 1);
          O[(((size_t)(b * HH + h)) * LL + l) * DD + d] = (_Float16)v;
        } else {
          float* O = (float*)outp;
          O[(size_t)gm * N + gn] = v;
        }
      }
    }
  }
}

// ---------------------------------------------------------------------------
// Elementwise helpers
// ---------------------------------------------------------------------------
__global__ void f32_to_f16_kernel(const float* __restrict__ src,
                                  _Float16* __restrict__ dst, int n) {
  int i = blockIdx.x * 256 + threadIdx.x;
  if (i < n) dst[i] = (_Float16)src[i];
}
__global__ void transpose_w_kernel(const float* __restrict__ W,
                                   _Float16* __restrict__ Wt) {
  int i = blockIdx.x * 256 + threadIdx.x;  // output index [n][k]
  if (i >= HID * HID) return;
  const int nn = i >> 10, k = i & 1023;
  Wt[i] = (_Float16)W[(size_t)k * HID + nn];
}
__global__ void init_cent_kernel(const _Float16* __restrict__ Qh,
                                 float* __restrict__ centF) {
  int i = blockIdx.x * 256 + threadIdx.x;  // BH*CC*DD
  if (i >= BHD * CC * DD) return;
  const int bh = i / (CC * DD);
  const int rem = i - bh * (CC * DD);
  const int c = rem >> 6, d = rem & 63;
  centF[i] = (float)Qh[((size_t)bh * LL + c) * DD + d];
}

// ---------------------------------------------------------------------------
// k-means: assignment via WMMA dot products + per-row argmin
// ---------------------------------------------------------------------------
__global__ __launch_bounds__(256) void kmeans_assign_kernel(
    const _Float16* __restrict__ Qh, const float* __restrict__ centF,
    int* __restrict__ assign) {
  __shared__ _Float16 Qs[64][72];
  __shared__ _Float16 Cs[128][72];
  __shared__ float cnorm[128];
  __shared__ float dots[64][132];
  __shared__ float bestd[64];
  __shared__ int   besti[64];
  __shared__ float redd[64][4];
  __shared__ int   redi[64][4];

  const int t = threadIdx.x;
  const int bh = blockIdx.y;
  const int lBase = blockIdx.x * 64;
  const int warp = t >> 5, lane = t & 31, hf = lane >> 4, ln = lane & 15;
  const int m0 = (warp & 3) * 16;
  const int n0 = (warp >> 2) * 64;

  {
    const int row = t >> 2, seg = (t & 3) << 4;
    const _Float16* g = Qh + ((size_t)bh * LL + lBase + row) * DD + seg;
    async_copy16(g,     &Qs[row][seg]);
    async_copy16(g + 8, &Qs[row][seg + 8]);
  }
  if (t < 64) { bestd[t] = 3.0e38f; besti[t] = 0; }
  async_fence();
  __syncthreads();

  for (int ch = 0; ch < 2; ++ch) {
    {
      const int row = t >> 1, seg = (t & 1) << 5;
      const float* g = centF + ((size_t)bh * CC + ch * 128 + row) * DD + seg;
#pragma unroll
      for (int j = 0; j < 32; ++j) Cs[row][seg + j] = (_Float16)g[j];
    }
    __syncthreads();
    if (t < 128) {
      float s = 0.f;
#pragma unroll
      for (int d = 0; d < DD; ++d) { float c = (float)Cs[t][d]; s += c * c; }
      cnorm[t] = s;
    }
    __syncthreads();

    v8f acc[4];
#pragma unroll
    for (int j = 0; j < 4; ++j)
#pragma unroll
      for (int r = 0; r < 8; ++r) acc[j][r] = 0.f;
#pragma unroll
    for (int kk = 0; kk < DD; kk += 32) {
      v16h af = lda_frag(&Qs[m0 + ln][kk], hf);
#pragma unroll
      for (int j = 0; j < 4; ++j) {
        v16h bf = ldb_frag(&Cs[n0 + j * 16 + ln][kk], hf);
        acc[j] = wmma16(af, bf, acc[j]);
      }
    }
#pragma unroll
    for (int j = 0; j < 4; ++j)
#pragma unroll
      for (int r = 0; r < 8; ++r)
        dots[m0 + hf * 8 + r][n0 + j * 16 + ln] = acc[j][r];
    __syncthreads();

    {
      const int row = t >> 2, q = t & 3;
      float bd = 3.0e38f; int bi = 0;
#pragma unroll
      for (int ccx = 0; ccx < 32; ++ccx) {
        const int col = q * 32 + ccx;
        const float dist = cnorm[col] - 2.0f * dots[row][col];
        if (dist < bd) { bd = dist; bi = ch * 128 + col; }
      }
      redd[row][q] = bd; redi[row][q] = bi;
    }
    __syncthreads();
    if (t < 64) {
      float bd = bestd[t]; int bi = besti[t];
#pragma unroll
      for (int q = 0; q < 4; ++q)
        if (redd[t][q] < bd) { bd = redd[t][q]; bi = redi[t][q]; }
      bestd[t] = bd; besti[t] = bi;
    }
    __syncthreads();
  }
  if (t < 64) assign[(size_t)bh * LL + lBase + t] = besti[t];
}

__global__ void kmeans_zero_kernel(float* sums, float* counts, int nsum, int ncnt) {
  int i = blockIdx.x * 256 + threadIdx.x;
  if (i < nsum) sums[i] = 0.f;
  if (i < ncnt) counts[i] = 0.f;
}
__global__ void kmeans_accum_kernel(const _Float16* __restrict__ Qh,
                                    const int* __restrict__ assign,
                                    float* __restrict__ sums,
                                    float* __restrict__ counts) {
  int i = blockIdx.x * 256 + threadIdx.x;  // over BH*LL
  if (i >= BHD * LL) return;
  const int bh = i >> 12;
  const int c = assign[i];
  atomicAdd(&counts[bh * CC + c], 1.0f);
  const _Float16* q = Qh + (size_t)i * DD;
  float* s = sums + ((size_t)bh * CC + c) * DD;
#pragma unroll
  for (int d = 0; d < DD; ++d) atomicAdd(&s[d], (float)q[d]);
}
__global__ void kmeans_newcent_kernel(const float* __restrict__ sums,
                                      const float* __restrict__ counts,
                                      float* __restrict__ centF, int n) {
  int i = blockIdx.x * 256 + threadIdx.x;
  if (i >= n) return;
  const float cnt = counts[i >> 6];
  if (cnt > 0.f) centF[i] = sums[i] / cnt;
}

// ---------------------------------------------------------------------------
// Flash-style clustered attention: Oc = softmax(cent K^T * scale) V
// One block per (bh, 64 centroid rows); streams over L in 64-wide tiles.
// ---------------------------------------------------------------------------
__global__ __launch_bounds__(256) void attn_kernel(
    const float* __restrict__ centF, const _Float16* __restrict__ Kh,
    const _Float16* __restrict__ Vh, const unsigned char* __restrict__ lmask,
    _Float16* __restrict__ Och) {
  __shared__ _Float16 Ca[64][72];
  __shared__ _Float16 Kt[64][72];   // [l][d] == B^T for cent*K^T
  __shared__ _Float16 Vt[64][72];   // [d][l] == B^T for P*V
  __shared__ float    Sm[64][68];
  __shared__ _Float16 Pm[64][72];
  __shared__ float mrow[64], lrow[64], scrow[64];
  __shared__ float red[64][4];
  __shared__ unsigned char mv[64];

  const int t = threadIdx.x;
  const int bh = blockIdx.y;
  const int b = bh / HH;
  const int cBase = blockIdx.x * 64;
  const int warp = t >> 5, lane = t & 31, hf = lane >> 4, ln = lane & 15;
  const int m0 = (warp & 3) * 16;
  const int n0 = (warp >> 2) * 32;
  const float scale = 0.125f;  // 1/sqrt(64)

  {
    const int row = t >> 2, seg = (t & 3) << 4;
    const float* g = centF + ((size_t)bh * CC + cBase + row) * DD + seg;
#pragma unroll
    for (int j = 0; j < 16; ++j) Ca[row][seg + j] = (_Float16)g[j];
  }
  if (t < 64) { mrow[t] = -3.0e38f; lrow[t] = 0.f; }
  v8f oacc[2];
#pragma unroll
  for (int j = 0; j < 2; ++j)
#pragma unroll
    for (int r = 0; r < 8; ++r) oacc[j][r] = 0.f;
  __syncthreads();

  for (int lb = 0; lb < LL; lb += 64) {
    {
      const int row = t >> 2, seg = (t & 3) << 4;
      const _Float16* gk = Kh + ((size_t)bh * LL + lb + row) * DD + seg;
      async_copy16(gk,     &Kt[row][seg]);       // ASYNCcnt path
      async_copy16(gk + 8, &Kt[row][seg + 8]);
      const _Float16* gv = Vh + ((size_t)bh * LL + lb + row) * DD + seg;
      v8h v0 = *(const v8h*)gv, v1 = *(const v8h*)(gv + 8);
#pragma unroll
      for (int j = 0; j < 8; ++j) {
        Vt[seg + j][row] = v0[j];
        Vt[seg + 8 + j][row] = v1[j];
      }
    }
    if (t < 64) mv[t] = lmask[(size_t)b * LL + lb + t];
    async_fence();
    __syncthreads();

    v8f sacc[2];
#pragma unroll
    for (int j = 0; j < 2; ++j)
#pragma unroll
      for (int r = 0; r < 8; ++r) sacc[j][r] = 0.f;
#pragma unroll
    for (int kk = 0; kk < DD; kk += 32) {
      v16h af = lda_frag(&Ca[m0 + ln][kk], hf);
#pragma unroll
      for (int j = 0; j < 2; ++j) {
        v16h bf = ldb_frag(&Kt[n0 + j * 16 + ln][kk], hf);
        sacc[j] = wmma16(af, bf, sacc[j]);
      }
    }
#pragma unroll
    for (int j = 0; j < 2; ++j)
#pragma unroll
      for (int r = 0; r < 8; ++r)
        Sm[m0 + hf * 8 + r][n0 + j * 16 + ln] = sacc[j][r] * scale;
    __syncthreads();

    {  // row max over this tile
      const int row = t >> 2, q = t & 3;
      float mx = -3.0e38f;
#pragma unroll
      for (int ccx = 0; ccx < 16; ++ccx) {
        const int col = q * 16 + ccx;
        mx = fmaxf(mx, mv[col] ? Sm[row][col] : NEGV);
      }
      red[row][q] = mx;
    }
    __syncthreads();
    if (t < 64) {
      const float tm = fmaxf(fmaxf(red[t][0], red[t][1]), fmaxf(red[t][2], red[t][3]));
      const float mn = fmaxf(mrow[t], tm);
      scrow[t] = __expf(mrow[t] - mn);
      mrow[t] = mn;
    }
    __syncthreads();
    {  // exp + partial sums, store P as f16
      const int row = t >> 2, q = t & 3;
      const float mn = mrow[row];
      float s = 0.f;
#pragma unroll
      for (int ccx = 0; ccx < 16; ++ccx) {
        const int col = q * 16 + ccx;
        const float v = mv[col] ? Sm[row][col] : NEGV;
        const float p = __expf(v - mn);
        Pm[row][col] = (_Float16)p;
        s += p;
      }
      red[row][q] = s;
    }
    __syncthreads();
    if (t < 64)
      lrow[t] = lrow[t] * scrow[t] + (red[t][0] + red[t][1] + red[t][2] + red[t][3]);
#pragma unroll
    for (int j = 0; j < 2; ++j)
#pragma unroll
      for (int r = 0; r < 8; ++r) oacc[j][r] *= scrow[m0 + hf * 8 + r];
#pragma unroll
    for (int kk = 0; kk < 64; kk += 32) {
      v16h af = lda_frag(&Pm[m0 + ln][kk], hf);
#pragma unroll
      for (int j = 0; j < 2; ++j) {
        v16h bf = ldb_frag(&Vt[n0 + j * 16 + ln][kk], hf);
        oacc[j] = wmma16(af, bf, oacc[j]);
      }
    }
    __syncthreads();
  }
#pragma unroll
  for (int j = 0; j < 2; ++j) {
#pragma unroll
    for (int r = 0; r < 8; ++r) {
      const int crow = m0 + hf * 8 + r;
      const int d = n0 + j * 16 + ln;
      const float denom = fmaxf(lrow[crow], 1.0e-30f);
      Och[((size_t)bh * CC + cBase + crow) * DD + d] = (_Float16)(oacc[j][r] / denom);
    }
  }
}

__global__ void gather_kernel(const _Float16* __restrict__ Och,
                              const int* __restrict__ assign,
                              _Float16* __restrict__ Oh, int n) {
  int i = blockIdx.x * 256 + threadIdx.x;
  if (i >= n) return;
  const int bl = i >> 10, col = i & 1023;
  const int b = bl >> 12, l = bl & (LL - 1);
  const int h = col >> 6, d = col & 63;
  const int bh = b * HH + h;
  const int c = assign[(size_t)bh * LL + l];
  Oh[i] = Och[((size_t)bh * CC + c) * DD + d];
}

// ---------------------------------------------------------------------------
extern "C" void kernel_launch(void* const* d_in, const int* in_sizes, int n_in,
                              void* d_out, int out_size, void* d_ws, size_t ws_size,
                              hipStream_t stream) {
  (void)in_sizes; (void)n_in; (void)out_size; (void)ws_size;
  const float* X = (const float*)d_in[0];
  const unsigned char* amask = (const unsigned char*)d_in[1];
  const unsigned char* lmask = (const unsigned char*)d_in[2];
  const float* Wq = (const float*)d_in[3];
  const float* bq = (const float*)d_in[4];
  const float* Wk = (const float*)d_in[5];
  const float* bk = (const float*)d_in[6];
  const float* Wv = (const float*)d_in[7];
  const float* bv = (const float*)d_in[8];
  const float* Wo = (const float*)d_in[9];
  const float* bo = (const float*)d_in[10];
  (void)amask;

  char* ws = (char*)d_ws;
  size_t off = 0;
  auto alloc = [&](size_t bytes) -> void* {
    off = (off + 255) & ~(size_t)255;
    void* p = ws + off;
    off += bytes;
    return p;
  };
  const size_t nX = (size_t)BB * LL * HID;       // 8M
  _Float16* Xh   = (_Float16*)alloc(nX * 2);
  _Float16* WqT  = (_Float16*)alloc((size_t)HID * HID * 2);
  _Float16* WkT  = (_Float16*)alloc((size_t)HID * HID * 2);
  _Float16* WvT  = (_Float16*)alloc((size_t)HID * HID * 2);
  _Float16* WoT  = (_Float16*)alloc((size_t)HID * HID * 2);
  _Float16* Qh   = (_Float16*)alloc(nX * 2);
  _Float16* Kh   = (_Float16*)alloc(nX * 2);
  _Float16* Vh   = (_Float16*)alloc(nX * 2);
  float*    centF  = (float*)alloc((size_t)BHD * CC * DD * 4);
  float*    sums   = (float*)alloc((size_t)BHD * CC * DD * 4);
  float*    counts = (float*)alloc((size_t)BHD * CC * 4);
  int*      assign = (int*)alloc((size_t)BHD * LL * 4);
  _Float16* Och    = (_Float16*)alloc((size_t)BHD * CC * DD * 2);
  _Float16* Oh     = (_Float16*)alloc(nX * 2);

  // 1. precision conversion + weight transposes
  f32_to_f16_kernel<<<(nX + 255) / 256, 256, 0, stream>>>(X, Xh, (int)nX);
  const int nW = HID * HID;
  transpose_w_kernel<<<(nW + 255) / 256, 256, 0, stream>>>(Wq, WqT);
  transpose_w_kernel<<<(nW + 255) / 256, 256, 0, stream>>>(Wk, WkT);
  transpose_w_kernel<<<(nW + 255) / 256, 256, 0, stream>>>(Wv, WvT);
  transpose_w_kernel<<<(nW + 255) / 256, 256, 0, stream>>>(Wo, WoT);

  // 2. Q/K/V projections (WMMA GEMM, fused permute to [B,H,L,D] f16)
  const int M = BB * LL;
  dim3 gg(HID / 128, M / 128);
  gemm_kernel<0><<<gg, 256, 0, stream>>>(Xh, WqT, bq, Qh, M, HID, HID);
  gemm_kernel<0><<<gg, 256, 0, stream>>>(Xh, WkT, bk, Kh, M, HID, HID);
  gemm_kernel<0><<<gg, 256, 0, stream>>>(Xh, WvT, bv, Vh, M, HID, HID);

  // 3. k-means: init centroids = first C queries, then 10 scan steps + final
  const int nC = BHD * CC * DD;
  init_cent_kernel<<<(nC + 255) / 256, 256, 0, stream>>>(Qh, centF);
  for (int it = 0; it < 11; ++it) {
    kmeans_assign_kernel<<<dim3(LL / 64, BHD), 256, 0, stream>>>(Qh, centF, assign);
    kmeans_zero_kernel<<<(nC + 255) / 256, 256, 0, stream>>>(sums, counts, nC, BHD * CC);
    kmeans_accum_kernel<<<(BHD * LL + 255) / 256, 256, 0, stream>>>(Qh, assign, sums, counts);
    kmeans_newcent_kernel<<<(nC + 255) / 256, 256, 0, stream>>>(sums, counts, centF, nC);
  }

  // 4. clustered attention (flash-style WMMA) -> Oc per centroid
  attn_kernel<<<dim3(CC / 64, BHD), 256, 0, stream>>>(centF, Kh, Vh, lmask, Och);

  // 5. gather per-token outputs by assignment, then output projection
  gather_kernel<<<(nX + 255) / 256, 256, 0, stream>>>(Och, assign, Oh, (int)nX);
  gemm_kernel<1><<<gg, 256, 0, stream>>>(Oh, WoT, bo, d_out, M, HID, HID);
}
